// CenterLoss_79345225826748
// MI455X (gfx1250) — matrix operand order repaired and checked by reading, not compiled, not run on hardware
//
#include <hip/hip_runtime.h>

typedef __attribute__((ext_vector_type(16))) _Float16 v16h;
typedef __attribute__((ext_vector_type(8)))  float    v8f;

#define BATCH 8
#define NTOK  4096
#define CH    256

// workspace layout (float offsets)
#define WS_POS   0      // [BATCH*CH] sum(m*E)  -> pos_center after finalize
#define WS_NEG   2048   // [BATCH*CH] sum(E)    -> neg_center after finalize
#define WS_CNT   4096   // [BATCH]    sum(ep mask)
#define WS_PP    4104   // [BATCH]    |pos_center|^2
#define WS_NN    4112   // [BATCH]    |neg_center|^2
#define WS_ACCP  4120   // [BATCH]    sum sq_pos * q_m
#define WS_ACCN  4128   // [BATCH]    sum sq_neg * (1-q_m)
#define WS_QCNT  4136   // [BATCH]    sum q_m
#define WS_POSH  4144   // [BATCH*CH f16 = BATCH*128 floats] packed f16 pos centers
#define WS_NEGH  5168   // [BATCH*CH f16]
#define WS_TOTAL 6192

__device__ __forceinline__ size_t mask_idx(int b, int n) {
    // mask[:,0,::16,::16] flattened: n = row*64 + col
    return (size_t)b * 1048576 + (size_t)(n >> 6) * 16384 + (size_t)(n & 63) * 16;
}

__global__ void k_zero(float* ws) {
    int i = blockIdx.x * 256 + threadIdx.x;
    if (i < WS_TOTAL) ws[i] = 0.f;
}

// Phase 1: per (batch, 256-token chunk) block; thread = channel.
// Accumulates sum(E) and sum(m*E) over tokens with coalesced loads.
__global__ void k_centers_accum(const float* __restrict__ emb,
                                const float* __restrict__ mask,
                                float* __restrict__ ws) {
    __shared__ float mv[256];
    int b = blockIdx.x >> 4;
    int chunk = blockIdx.x & 15;
    int t = threadIdx.x;          // channel
    int n0 = chunk * 256;
    mv[t] = mask[mask_idx(b, n0 + t)];
    __syncthreads();
    const float* ep = emb + ((size_t)b * NTOK + n0) * CH + t;
    float accA = 0.f, accP = 0.f, accM = 0.f;
#pragma unroll 8
    for (int i = 0; i < 256; ++i) {
        float e = ep[(size_t)i * CH];
        float m = mv[i];
        accA += e;
        accP += m * e;
        accM += m;
    }
    atomicAdd(&ws[WS_POS + b * CH + t], accP);
    atomicAdd(&ws[WS_NEG + b * CH + t], accA);
    if (t == 0) atomicAdd(&ws[WS_CNT + b], accM);
}

// Phase 1b: centers, squared norms, and f16-packed copies for WMMA B-tiles.
__global__ void k_finalize_centers(float* __restrict__ ws) {
    __shared__ float sp[256], sn[256];
    int b = blockIdx.x, t = threadIdx.x;
    float pn  = ws[WS_POS + b * CH + t];
    float al  = ws[WS_NEG + b * CH + t];
    float cnt = ws[WS_CNT + b];
    float pc = pn / (cnt + 0.1f);
    float nc = (al - pn) / ((float)NTOK - cnt + 0.1f);
    ws[WS_POS + b * CH + t] = pc;
    ws[WS_NEG + b * CH + t] = nc;
    _Float16* ph = (_Float16*)(ws + WS_POSH) + b * CH;
    _Float16* nh = (_Float16*)(ws + WS_NEGH) + b * CH;
    ph[t] = (_Float16)pc;
    nh[t] = (_Float16)nc;
    sp[t] = pc * pc;
    sn[t] = nc * nc;
    __syncthreads();
    for (int s = 128; s > 0; s >>= 1) {
        if (t < s) { sp[t] += sp[t + s]; sn[t] += sn[t + s]; }
        __syncthreads();
    }
    if (t == 0) { ws[WS_PP + b] = sp[0]; ws[WS_NN + b] = sn[0]; }
}

// Phase 2: per-token dots with both centers via V_WMMA_F32_16X16X32_F16.
// Each wave owns 16 tokens; B holds pos_center (col 0) and neg_center (col 1).
// All 8 B-fragments are staged in LDS once per block -> uniform inner loop.
__global__ void k_query_loss(const float* __restrict__ qemb,
                             const float* __restrict__ qmask,
                             float* ws) {
    __shared__ v16h bts[256];          // [tile][lane] B fragments, 8 KB
    __shared__ float dotP[128], dotN[128];
    int b     = blockIdx.x >> 5;
    int chunk = blockIdx.x & 31;
    int wave  = threadIdx.x >> 5;
    int lane  = threadIdx.x & 31;
    int tokbase = chunk * 128 + wave * 16;
    int m = lane & 15;     // my A-matrix row (token within tile) / B column
    int g = lane >> 4;     // lane-group selects K half-sets

    // Cooperative B-fragment staging: thread = (tile, lane) pair.
    {
        int tile = threadIdx.x >> 5;
        int l    = threadIdx.x & 31;
        int col  = l & 15;
        int gg   = l >> 4;
        v16h v = {};
        if (col < 2) {
            const _Float16* src =
                ((col == 0) ? (const _Float16*)(ws + WS_POSH)
                            : (const _Float16*)(ws + WS_NEGH))
                + b * CH + tile * 32 + 16 * gg;
            v = *(const v16h*)src;
        }
        bts[threadIdx.x] = v;
    }
    __syncthreads();

    const float* qrow = qemb + ((size_t)(b * NTOK + tokbase + m)) * CH;
    if (lane < 16) __builtin_prefetch(&qmask[mask_idx(b, tokbase + lane)], 0, 0);

    v8f acc = {};
    float qq = 0.f;
#pragma unroll 2
    for (int t = 0; t < 8; ++t) {
        int c0 = t * 32;
        // A tile: lanes 0-15 hold K {0..15}, lanes 16-31 hold K {16..31}
        const float* p1 = qrow + c0 + 16 * g;
        const float* p2 = p1 + 8;
        v16h a;
#pragma unroll
        for (int j = 0; j < 8; ++j) {
            float x = p1[j], y = p2[j];
            a[j]     = (_Float16)x;
            a[8 + j] = (_Float16)y;
            qq += x * x + y * y;
        }
        v16h bt = bts[t * 32 + lane];
        acc = __builtin_amdgcn_wmma_f32_16x16x32_f16(
            false, a, false, bt, (short)0, acc, false, false);
    }
    // full-row |q|^2: combine the two K half-sets
    qq += __shfl_xor(qq, 16, 32);

    // D layout: D(r, col) in VGPR r at lane col (rows 0-7), lane 16+col (rows 8-15)
    if (lane == 0)  { for (int r = 0; r < 8; ++r) dotP[wave * 16 + r]     = acc[r]; }
    if (lane == 16) { for (int r = 0; r < 8; ++r) dotP[wave * 16 + 8 + r] = acc[r]; }
    if (lane == 1)  { for (int r = 0; r < 8; ++r) dotN[wave * 16 + r]     = acc[r]; }
    if (lane == 17) { for (int r = 0; r < 8; ++r) dotN[wave * 16 + 8 + r] = acc[r]; }
    __syncthreads();

    float cp = 0.f, cn = 0.f, cq = 0.f;
    if (lane < 16) {
        int n = tokbase + lane;
        float dp = dotP[wave * 16 + lane];
        float dn = dotN[wave * 16 + lane];
        float pp = ws[WS_PP + b];
        float nn = ws[WS_NN + b];
        float sqp = qq - 2.f * dp + pp;   // |q - pos_center|^2
        float sqn = qq - 2.f * dn + nn;   // |q - neg_center|^2
        float qm = qmask[mask_idx(b, n)];
        cp = sqp * qm;
        cn = sqn * (1.f - qm);
        cq = qm;
    }
    for (int s = 8; s > 0; s >>= 1) {
        cp += __shfl_xor(cp, s, 32);
        cn += __shfl_xor(cn, s, 32);
        cq += __shfl_xor(cq, s, 32);
    }
    if (lane == 0) {
        atomicAdd(&ws[WS_ACCP + b], cp);
        atomicAdd(&ws[WS_ACCN + b], cn);
        atomicAdd(&ws[WS_QCNT + b], cq);
    }
}

__global__ void k_final(const float* __restrict__ ws, float* __restrict__ out) {
    int lane = threadIdx.x;
    float pl = 0.f, nl = 0.f;
    if (lane < BATCH) {
        float accp = ws[WS_ACCP + lane];
        float accn = ws[WS_ACCN + lane];
        float qc   = ws[WS_QCNT + lane];
        float npos = qc;
        float nneg = (float)NTOK - qc;
        pl = (npos > 0.f) ? accp / (fmaxf(npos, 1.f) * (float)CH) : 0.f;
        nl = (nneg > 0.f) ? accn / (fmaxf(nneg, 1.f) * (float)CH) : 0.f;
    }
    for (int s = 4; s > 0; s >>= 1) {
        pl += __shfl_xor(pl, s, 32);
        nl += __shfl_xor(nl, s, 32);
    }
    if (lane == 0) {
        out[0] = (pl + nl) / (float)BATCH;
        out[1] = pl / (float)BATCH;
        out[2] = nl / (float)BATCH;
    }
}

extern "C" void kernel_launch(void* const* d_in, const int* in_sizes, int n_in,
                              void* d_out, int out_size, void* d_ws, size_t ws_size,
                              hipStream_t stream) {
    const float* ep_emb  = (const float*)d_in[0];
    const float* ep_mask = (const float*)d_in[1];
    const float* q_emb   = (const float*)d_in[2];
    const float* q_mask  = (const float*)d_in[3];
    float* ws  = (float*)d_ws;
    float* out = (float*)d_out;

    k_zero<<<(WS_TOTAL + 255) / 256, 256, 0, stream>>>(ws);
    k_centers_accum<<<BATCH * 16, 256, 0, stream>>>(ep_emb, ep_mask, ws);
    k_finalize_centers<<<BATCH, 256, 0, stream>>>(ws);
    k_query_loss<<<BATCH * 32, 256, 0, stream>>>(q_emb, q_mask, ws);
    k_final<<<1, 32, 0, stream>>>(ws, out);
}
